// SRNLayer_55662776156232
// MI455X (gfx1250) — matrix operand order repaired and checked by reading, not compiled, not run on hardware
//
#include <hip/hip_runtime.h>
#include <math.h>

// ---------------------------------------------------------------------------
// SRN layer, MI455X (gfx1250, wave32, WMMA).
// err(b,o,l) = (1/N)(||y_hat||^2 - ||G_o^T y_hat||^2),  G_o = X_o L_o^{-T},
// X_o^T X_o = L_o L_o^T (Cholesky).  Output exp(-err).  N=576, P=9, O=16.
// U(144x16) = Gall(144x576,bf16) x Ytile(576x16,bf16) via wmma f32 acc.
// ---------------------------------------------------------------------------

typedef __attribute__((ext_vector_type(16))) __bf16 v16bf;
typedef __attribute__((ext_vector_type(8)))  float  v8f;

#define KSZ    3
#define C_IN   64
#define N_FEAT 576          // C * k * k
#define O_OUT  16
#define PDEG   9            // [1, w0, w1, ..., w0^4, w1^4]
#define ROWS   144          // O_OUT * PDEG
#define HW     32
#define TILE_N 16           // l-columns per block
#define KSTEPS 18           // 576 / 32
#define NPAIRS 288          // 576 / 2 packed bf16 pairs

__device__ __forceinline__ unsigned f32_to_bf16_rne(float f) {
    union { float f; unsigned u; } v; v.f = f;
    unsigned r = v.u + 0x7FFFu + ((v.u >> 16) & 1u);
    return r >> 16;
}

__device__ __forceinline__ float load_y(const float* __restrict__ y2,
                                        const float* __restrict__ noise,
                                        int b, int n, int l, int h, int w) {
    int c = n / (KSZ * KSZ), rem = n - c * (KSZ * KSZ);
    int ki = rem / KSZ, kj = rem - ki * KSZ;
    int hh = h + ki - 1, ww = w + kj - 1;
    float v = 0.f;
    if ((unsigned)hh < (unsigned)HW && (unsigned)ww < (unsigned)HW)
        v = y2[((b * C_IN + c) * HW + hh) * HW + ww];
    return v + 1e-4f * noise[((b * N_FEAT + n) << 10) + l];
}

// ---------------------------------------------------------------------------
// Prep: per output channel o, build X (576x9), Gram, Cholesky, G = X L^{-T},
// store Gall as bf16 row-major (144 x 576) into workspace.
// ---------------------------------------------------------------------------
__global__ __launch_bounds__(128)
void srn_prep_kernel(const float* __restrict__ weights,      // (16,64,3,3,2)
                     unsigned short* __restrict__ Gall)      // (144,576) bf16
{
    __shared__ float Xl[N_FEAT * PDEG];
    __shared__ float Cov[PDEG * PDEG];
    __shared__ float Lm[PDEG * PDEG];
    const int o = blockIdx.x;
    const int t = threadIdx.x;

    for (int e = t; e < N_FEAT * PDEG; e += 128) {
        int n = e / PDEG, p = e - n * PDEG;
        float val = 1.0f;
        if (p != 0) {
            int d   = (p - 1) / 2 + 1;
            int s   = (p - 1) & 1;
            int c   = n / (KSZ * KSZ);
            int rem = n - c * (KSZ * KSZ);
            int ki  = rem / KSZ, kj = rem - ki * KSZ;
            float w = weights[(((o * C_IN + c) * KSZ + ki) * KSZ + kj) * 2 + s];
            val = w;
            for (int dd = 1; dd < d; ++dd) val *= w;
        }
        Xl[e] = val;
    }
    __syncthreads();

    if (t < PDEG * PDEG) {                      // Gram matrix X^T X
        int p = t / PDEG, q = t - p * PDEG;
        float acc = 0.f;
        for (int n = 0; n < N_FEAT; ++n) acc += Xl[n * PDEG + p] * Xl[n * PDEG + q];
        Cov[t] = acc;
    }
    __syncthreads();

    if (t == 0) {                               // Cholesky (9x9)
        for (int p = 0; p < PDEG; ++p)
            for (int q = 0; q < PDEG; ++q)
                Lm[p * PDEG + q] = (q <= p) ? Cov[p * PDEG + q] : 0.f;
        for (int k = 0; k < PDEG; ++k) {
            float d = sqrtf(Lm[k * PDEG + k]);
            Lm[k * PDEG + k] = d;
            float inv = 1.f / d;
            for (int r = k + 1; r < PDEG; ++r) Lm[r * PDEG + k] *= inv;
            for (int r = k + 1; r < PDEG; ++r)
                for (int c = k + 1; c <= r; ++c)
                    Lm[r * PDEG + c] -= Lm[r * PDEG + k] * Lm[c * PDEG + k];
        }
    }
    __syncthreads();

    for (int n = t; n < N_FEAT; n += 128) {     // G row n: solve L g = x_n
        float g[PDEG];
        for (int p = 0; p < PDEG; ++p) {
            float acc = Xl[n * PDEG + p];
            for (int q = 0; q < p; ++q) acc -= Lm[p * PDEG + q] * g[q];
            g[p] = acc / Lm[p * PDEG + p];
        }
        for (int p = 0; p < PDEG; ++p)
            Gall[(o * PDEG + p) * N_FEAT + n] = (unsigned short)f32_to_bf16_rne(g[p]);
    }
}

// ---------------------------------------------------------------------------
// Main: one block = (batch b, 16 l-columns).  288 threads = 9 waves.
// Staging pass: thread t owns column (t&15) and n-chunk (t>>4)*32..+31;
// materializes unfold+noise, accumulates s1/s2 in registers, stores the
// chunk pre-packed as bf16 pairs in plain k order.  Wave wv then runs
// the 18-step wmma_f32_16x16x32_bf16 loop for rows 16wv..16wv+15 of Gall.
// ---------------------------------------------------------------------------
union BfFrag { unsigned u[8]; v16bf v; };

__global__ __launch_bounds__(288)
void srn_main_kernel(const float* __restrict__ y2,      // (8,64,32,32)
                     const float* __restrict__ noise,   // (8,576,1024)
                     const unsigned short* __restrict__ Gall, // (144,576) bf16
                     float* __restrict__ out)           // (8,16,32,32)
{
    __shared__ unsigned ybf [NPAIRS * TILE_N];   // 18 KB, packed bf16 pairs
    __shared__ float    gbuf[ROWS * TILE_N];     //  9 KB
    __shared__ float    psum[KSTEPS * TILE_N];
    __shared__ float    psq [KSTEPS * TILE_N];
    __shared__ float    sfin[TILE_N];            // s2 per column
    __shared__ float    vfin[TILE_N];            // 1/(576*var) per column

    const int b  = blockIdx.x >> 6;
    const int l0 = (blockIdx.x & 63) * TILE_N;
    const int t  = threadIdx.x;
    const int lane = t & 31;
    const int wv   = t >> 5;                   // 0..8 : row-tile of Gall

    // ---- fused staging: unfold + noise, s1/s2, bf16 pack -----------------
    {
        const int col = t & 15, chunk = t >> 4;       // 18 chunks x 16 cols
        const int l = l0 + col;
        const int h = l >> 5, w = l & 31;
        float s = 0.f, q = 0.f;
        for (int kp = 0; kp < 16; ++kp) {
            int n = chunk * 32 + 2 * kp;
            float y0 = load_y(y2, noise, b, n,     l, h, w);
            float y1 = load_y(y2, noise, b, n + 1, l, h, w);
            s += y0 + y1;
            q += y0 * y0 + y1 * y1;
            ybf[(chunk * 16 + kp) * TILE_N + col] =
                f32_to_bf16_rne(y0) | (f32_to_bf16_rne(y1) << 16);
        }
        psum[chunk * TILE_N + col] = s;
        psq [chunk * TILE_N + col] = q;
    }
    __syncthreads();

    // ---- WMMA k-loop: wave wv -> rows 16wv..16wv+15 ----------------------
    const int M    = lane & 15;                // A: row-in-tile, B/D: column
    const int half = lane >> 4;
    const unsigned* Gw = (const unsigned*)Gall;
    const int rowA = 16 * wv + M;
    v8f acc = {};
#pragma unroll 3
    for (int kk = 0; kk < KSTEPS; ++kk) {
        BfFrag a, bm;
        const int kbase = kk * 32;
#pragma unroll
        for (int v = 0; v < 8; ++v) {
            // A (16x32 bf16): vgpr v holds K pair {2v | 16+2(v-4)} + 8*half
            int ka = kbase + ((v < 4) ? (2 * v) : (16 + 2 * (v - 4))) + 8 * half;
            a.u[v] = Gw[(rowA * N_FEAT + ka) >> 1];
            // B (32x16 bf16): vgpr v holds K pair {2v | 16+2(v-4)} + 8*half; lane = N
            bm.u[v] = ybf[(kk * 16 + ((v < 4) ? v : (v + 4)) + 4 * half) * TILE_N + M];
        }
        acc = __builtin_amdgcn_wmma_f32_16x16x32_bf16(false, a.v, false, bm.v,
                                                      (short)0, acc, false, false);
    }
    // D layout: lane N = lane&15, vgpr v -> M = v + 8*half
#pragma unroll
    for (int v = 0; v < 8; ++v)
        gbuf[(16 * wv + v + 8 * half) * TILE_N + M] = acc[v];
    __syncthreads();

    // ---- per-column statistics (once per column) -------------------------
    if (t < TILE_N) {
        float s1 = 0.f, s2 = 0.f;
        for (int ch = 0; ch < KSTEPS; ++ch) {
            s1 += psum[ch * TILE_N + t];
            s2 += psq [ch * TILE_N + t];
        }
        float var = (s2 - s1 * s1 * (1.0f / 576.0f)) * (1.0f / 575.0f);
        sfin[t] = s2;
        vfin[t] = 1.0f / (576.0f * var);
    }
    __syncthreads();

    // ---- finalize: err = (s2 - ||u||^2) / (576*var), out = exp(-err) -----
    if (t < O_OUT * TILE_N) {
        int o = t >> 4, col = t & 15;
        float sumg2 = 0.f;
#pragma unroll
        for (int p = 0; p < PDEG; ++p) {
            float g = gbuf[(o * PDEG + p) * TILE_N + col];
            sumg2 += g * g;
        }
        float err = (sfin[col] - sumg2) * vfin[col];
        out[((b * O_OUT + o) << 10) + l0 + col] = expf(-err);
    }
}

// ---------------------------------------------------------------------------
extern "C" void kernel_launch(void* const* d_in, const int* in_sizes, int n_in,
                              void* d_out, int out_size, void* d_ws, size_t ws_size,
                              hipStream_t stream) {
    const float* y2      = (const float*)d_in[0];   // 8*64*32*32
    const float* weights = (const float*)d_in[1];   // 16*64*3*3*2
    const float* noise   = (const float*)d_in[2];   // 8*576*1024
    float* out = (float*)d_out;                     // 8*16*32*32

    unsigned short* Gall = (unsigned short*)d_ws;   // 144*576 bf16 = 165,888 B

    srn_prep_kernel<<<O_OUT, 128, 0, stream>>>(weights, Gall);
    srn_main_kernel<<<8 * (1024 / TILE_N), 288, 0, stream>>>(y2, noise, Gall, out);
}
